// CausalAttention_11665131176114
// MI455X (gfx1250) — matrix operand order repaired and checked by reading, not compiled, not run on hardware
//
#include <hip/hip_runtime.h>
#include <math.h>

// ---------------------------------------------------------------------------
// Causal attention for B=8, S=2048, D=1024 (fp32 in/out), MI455X / gfx1250.
// Strategy: bf16 WMMA (v_wmma_f32_16x16x32_bf16) everywhere, f32 accumulate.
//   Kernel 0: one-pass fp32 -> bf16 conversion of X and Wq/Wk/Wv (removes all
//             conversion VALU work and halves operand bytes in the GEMM loop).
//   Kernel 1: QKV projection GEMM (x @ W^T) on bf16 operands.
//             Q,K stored row-major bf16; V stored transposed Vt[b][d][s] so
//             the P@V B-operand is a contiguous 32B load per lane.
//   Kernel 2: flash-attention, 32-query tiles, online softmax, causal mask.
// ---------------------------------------------------------------------------

typedef __attribute__((ext_vector_type(16))) __bf16 v16bf;
typedef __attribute__((ext_vector_type(8)))  __bf16 v8bf;
typedef __attribute__((ext_vector_type(8)))  float  v8f;
typedef __attribute__((ext_vector_type(4)))  float  v4f;

namespace {
constexpr int Bsz  = 8;
constexpr int Seq  = 2048;
constexpr int Dim  = 1024;
constexpr int Ntok = Bsz * Seq;           // 16384
constexpr float Scale = 0.03125f;         // 1/sqrt(1024)
}

__device__ __forceinline__ v16bf bf_combine(v8bf lo, v8bf hi) {
  v16bf r;
#pragma unroll
  for (int e = 0; e < 8; ++e) { r[e] = lo[e]; r[8 + e] = hi[e]; }
  return r;
}

// ---------------------------------------------------------------------------
// Kernel 0: fp32 -> bf16, 8 elements per thread (emits v_cvt_pk_bf16_f32).
// n must be a multiple of 8 (true for all our arrays).
// ---------------------------------------------------------------------------
__global__ __launch_bounds__(256, 1)
void cvt_bf16_kernel(const float* __restrict__ src, __bf16* __restrict__ dst, int n) {
  const int idx = (blockIdx.x * 256 + threadIdx.x) * 8;
  if (idx >= n) return;
  v4f a = *(const v4f*)(src + idx);
  v4f b = *(const v4f*)(src + idx + 4);
  v8bf o;
#pragma unroll
  for (int e = 0; e < 4; ++e) {
    o[e]     = (__bf16)a[e];
    o[4 + e] = (__bf16)b[e];
  }
  *(v8bf*)(dst + idx) = o;
}

// ---------------------------------------------------------------------------
// Kernel 1: Y = Xb (Ntok x Dim, bf16) @ Wb^T (Wb row-major bf16), NT GEMM.
// Block: 256 threads = 8 waves (4 x 2). Block tile 128x128, wave tile 32x64.
// blockIdx.z selects {Wq->Q, Wk->K, Wv->Vt(transposed store)}.
// ---------------------------------------------------------------------------
__global__ __launch_bounds__(256, 1)
void qkv_gemm_kernel(const __bf16* __restrict__ Xb,
                     const __bf16* __restrict__ Wb,   // 3 * Dim * Dim, concat
                     __bf16* __restrict__ Q,
                     __bf16* __restrict__ K,
                     __bf16* __restrict__ Vt) {
  const int lane  = threadIdx.x & 31;
  const int wid   = threadIdx.x >> 5;      // 0..7
  const int wm    = wid & 3;               // 4 waves along M
  const int wn    = wid >> 2;              // 2 waves along N
  const int m0    = blockIdx.x * 128 + wm * 32;
  const int n0    = blockIdx.y * 128 + wn * 64;
  const int which = blockIdx.z;
  const __bf16* W = Wb + (size_t)which * Dim * Dim;

  const int lhalf = lane >> 4;             // 0/1
  const int lm    = lane & 15;

  // Per-lane base pointers (advance by 32 bf16 per k-step).
  const __bf16* pa0 = Xb + (size_t)(m0 + lm) * Dim + lhalf * 16;
  const __bf16* pa1 = pa0 + (size_t)16 * Dim;
  const __bf16* pb[4];
#pragma unroll
  for (int j = 0; j < 4; ++j)
    pb[j] = W + (size_t)(n0 + 16 * j + lm) * Dim + lhalf * 16;

  v8f acc[2][4] = {};

  for (int kk = 0; kk < Dim; kk += 32) {
    // A fragments: 16x32 bf16; lane = row, two contiguous 8-elem K chunks.
    v16bf a[2];
    a[0] = bf_combine(*(const v8bf*)(pa0 + kk), *(const v8bf*)(pa0 + kk + 8));
    a[1] = bf_combine(*(const v8bf*)(pa1 + kk), *(const v8bf*)(pa1 + kk + 8));
    // B fragments: 32x16 bf16; lane = column, 16 contiguous K.
    v16bf bfr[4];
#pragma unroll
    for (int j = 0; j < 4; ++j)
      bfr[j] = *(const v16bf*)(pb[j] + kk);

    // Prefetch next k-step's operands into cache.
    if (kk + 32 < Dim) {
      __builtin_prefetch(pa0 + kk + 32, 0, 1);
      __builtin_prefetch(pa1 + kk + 32, 0, 1);
      __builtin_prefetch(pb[0] + kk + 32, 0, 1);
      __builtin_prefetch(pb[2] + kk + 32, 0, 1);
    }

#pragma unroll
    for (int i = 0; i < 2; ++i)
#pragma unroll
      for (int j = 0; j < 4; ++j)
        acc[i][j] = __builtin_amdgcn_wmma_f32_16x16x32_bf16(
            false, a[i], false, bfr[j], (short)0, acc[i][j], false, false);
  }

  // Store. C/D layout: lane holds column n=lm; VGPR r -> row r + 8*lhalf.
  if (which < 2) {
    __bf16* Y = (which == 0) ? Q : K;
#pragma unroll
    for (int i = 0; i < 2; ++i)
#pragma unroll
      for (int j = 0; j < 4; ++j) {
        const int n = n0 + 16 * j + lm;
#pragma unroll
        for (int r = 0; r < 8; ++r) {
          const int m = m0 + 16 * i + r + 8 * lhalf;
          Y[(size_t)m * Dim + n] = (__bf16)acc[i][j][r];
        }
      }
  } else {
    // Transposed store: Vt[b][d][s]. (M-tile never crosses a batch: 2048%128==0.)
#pragma unroll
    for (int i = 0; i < 2; ++i)
#pragma unroll
      for (int j = 0; j < 4; ++j) {
        const int n = n0 + 16 * j + lm;                   // d index
#pragma unroll
        for (int r = 0; r < 8; ++r) {
          const int m = m0 + 16 * i + r + 8 * lhalf;      // token index
          const int b = m >> 11;
          const int s = m & (Seq - 1);
          Vt[(size_t)b * Dim * Seq + (size_t)n * Seq + s] = (__bf16)acc[i][j][r];
        }
      }
  }
}

// ---------------------------------------------------------------------------
// Kernel 2: causal flash attention. Block = 512 threads = 16 waves.
// Block handles 32 query rows of one batch; wave w owns d-slice [w*64, w*64+64).
// Per 32-key tile: partial scores per wave (over its d-slice) -> ds_add_f32
// reduce in LDS -> wave0 online softmax (causal mask) -> P(bf16) in LDS ->
// P @ Vt accumulates into per-wave O with running rescale.
// ---------------------------------------------------------------------------
__global__ __launch_bounds__(512, 1)
void flash_attn_kernel(const __bf16* __restrict__ Q,
                       const __bf16* __restrict__ K,
                       const __bf16* __restrict__ Vt,
                       float* __restrict__ Out) {
  __shared__ float  s_tile[32][33];
  __shared__ __bf16 p_tile[32][32];
  __shared__ float  alpha_sh[32];
  __shared__ float  l_sh[32];

  const int lane  = threadIdx.x & 31;
  const int wid   = threadIdx.x >> 5;      // 0..15
  const int lhalf = lane >> 4;
  const int lm    = lane & 15;
  const int q0    = blockIdx.x * 32;
  const int b     = blockIdx.y;
  const int ds0   = wid * 64;

  const __bf16* Qb = Q  + (size_t)b * Seq * Dim;
  const __bf16* Kb = K  + (size_t)b * Seq * Dim;
  const __bf16* Vb = Vt + (size_t)b * Dim * Seq;

  // Preload Q fragments for this wave's d-slice: [q-subtile][k-step of 32].
  v16bf qf[2][2];
#pragma unroll
  for (int i = 0; i < 2; ++i)
#pragma unroll
    for (int t = 0; t < 2; ++t) {
      const int row   = q0 + 16 * i + lm;
      const __bf16* p = Qb + (size_t)row * Dim + ds0 + 32 * t + lhalf * 16;
      qf[i][t] = bf_combine(*(const v8bf*)p, *(const v8bf*)(p + 8));
    }

  v8f o[2][4] = {};
  float m_run = -INFINITY;   // meaningful on wave 0, lane r = row r
  float l_run = 0.0f;

  const int nkt = q0 / 32 + 1;
  for (int kt = 0; kt < nkt; ++kt) {
    const int k0 = kt * 32;

    __syncthreads();
    for (int t = threadIdx.x; t < 32 * 32; t += 512)
      s_tile[t >> 5][t & 31] = 0.0f;
    __syncthreads();

    // ---- partial scores over this wave's d-slice: S = Q K^T ----
    v8f sacc[2][2] = {};
#pragma unroll
    for (int t = 0; t < 2; ++t) {
      v16bf kf[2];
#pragma unroll
      for (int jj = 0; jj < 2; ++jj) {
        const int key = k0 + 16 * jj + lm;
        kf[jj] = *(const v16bf*)(Kb + (size_t)key * Dim + ds0 + 32 * t + lhalf * 16);
      }
#pragma unroll
      for (int i = 0; i < 2; ++i)
#pragma unroll
        for (int jj = 0; jj < 2; ++jj)
          sacc[i][jj] = __builtin_amdgcn_wmma_f32_16x16x32_bf16(
              false, qf[i][t], false, kf[jj], (short)0, sacc[i][jj], false, false);
    }
    // Prefetch next key-tile's K rows while we reduce/softmax.
    if (k0 + 32 + lm < Seq) {
      __builtin_prefetch(Kb + (size_t)(k0 + 32 + lm) * Dim + ds0 + lhalf * 16, 0, 1);
    }
    // reduce partials across waves in LDS
#pragma unroll
    for (int i = 0; i < 2; ++i)
#pragma unroll
      for (int jj = 0; jj < 2; ++jj)
#pragma unroll
        for (int r = 0; r < 8; ++r)
          atomicAdd(&s_tile[16 * i + r + 8 * lhalf][16 * jj + lm], sacc[i][jj][r]);
    __syncthreads();

    // ---- online softmax update (wave 0, one row per lane) ----
    if (wid == 0) {
      const int r = lane;                         // query row in tile
      int nv = q0 + r - k0 + 1;                   // causal: keys k0..q0+r
      if (nv > 32) nv = 32;
      float mt = -INFINITY;
      for (int c = 0; c < nv; ++c)
        mt = fmaxf(mt, s_tile[r][c] * Scale);
      const float mn = fmaxf(m_run, mt);
      const float al = __expf(m_run - mn);
      float sum = 0.0f;
      for (int c = 0; c < 32; ++c) {
        float p = 0.0f;
        if (c < nv) p = __expf(s_tile[r][c] * Scale - mn);
        p_tile[r][c] = (__bf16)p;
        sum += p;
      }
      l_run = l_run * al + sum;
      m_run = mn;
      alpha_sh[r] = al;
    }
    __syncthreads();

    // ---- rescale O, then O += P @ V ----
#pragma unroll
    for (int i = 0; i < 2; ++i)
#pragma unroll
      for (int j = 0; j < 4; ++j)
#pragma unroll
        for (int r = 0; r < 8; ++r)
          o[i][j][r] *= alpha_sh[16 * i + r + 8 * lhalf];

    v16bf pf[2];
#pragma unroll
    for (int i = 0; i < 2; ++i) {
      const int row = 16 * i + lm;
      const int c0  = lhalf * 16;
      pf[i] = bf_combine(*(const v8bf*)&p_tile[row][c0],
                         *(const v8bf*)&p_tile[row][c0 + 8]);
    }
#pragma unroll
    for (int j = 0; j < 4; ++j) {
      const int d = ds0 + 16 * j + lm;
      v16bf vf = *(const v16bf*)(Vb + (size_t)d * Seq + k0 + lhalf * 16);
#pragma unroll
      for (int i = 0; i < 2; ++i)
        o[i][j] = __builtin_amdgcn_wmma_f32_16x16x32_bf16(
            false, pf[i], false, vf, (short)0, o[i][j], false, false);
    }
    __syncthreads();   // protect s_tile/p_tile before next iteration's zeroing
  }

  if (wid == 0) l_sh[lane] = l_run;
  __syncthreads();

  // ---- normalize and store fp32 ----
#pragma unroll
  for (int i = 0; i < 2; ++i)
#pragma unroll
    for (int j = 0; j < 4; ++j) {
      const int d = ds0 + 16 * j + lm;
#pragma unroll
      for (int r = 0; r < 8; ++r) {
        const int row = 16 * i + r + 8 * lhalf;
        const float inv = 1.0f / l_sh[row];
        Out[((size_t)b * Seq + q0 + row) * Dim + d] = o[i][j][r] * inv;
      }
    }
}

// ---------------------------------------------------------------------------
extern "C" void kernel_launch(void* const* d_in, const int* in_sizes, int n_in,
                              void* d_out, int out_size, void* d_ws, size_t ws_size,
                              hipStream_t stream) {
  (void)in_sizes; (void)n_in; (void)out_size; (void)ws_size;
  const float* X  = (const float*)d_in[0];
  const float* Wq = (const float*)d_in[1];
  const float* Wk = (const float*)d_in[2];
  const float* Wv = (const float*)d_in[3];
  float* Out = (float*)d_out;

  // Workspace layout (bf16 elements):
  //   [0        , 16M)  Q   (row-major)
  //   [16M      , 32M)  K   (row-major)
  //   [32M      , 48M)  Vt  (d-major, transposed V)
  //   [48M      , 64M)  Xb  (bf16 copy of X)
  //   [64M      , 67M)  Wb  (bf16 copies of Wq|Wk|Wv)
  // Total: 134 MB.
  const size_t qkv_elems = (size_t)Ntok * Dim;   // 16M
  const size_t w_elems   = (size_t)Dim * Dim;    // 1M
  __bf16* Qw = (__bf16*)d_ws;
  __bf16* Kw = Qw + qkv_elems;
  __bf16* Vt = Kw + qkv_elems;
  __bf16* Xb = Vt + qkv_elems;
  __bf16* Wb = Xb + qkv_elems;

  // Kernel 0: convert operands to bf16 once.
  {
    const int nX = (int)qkv_elems;            // 16,777,216
    const int nW = (int)w_elems;              // 1,048,576
    cvt_bf16_kernel<<<dim3(nX / (256 * 8)), dim3(256), 0, stream>>>(X, Xb, nX);
    cvt_bf16_kernel<<<dim3(nW / (256 * 8)), dim3(256), 0, stream>>>(Wq, Wb + 0 * w_elems, nW);
    cvt_bf16_kernel<<<dim3(nW / (256 * 8)), dim3(256), 0, stream>>>(Wk, Wb + 1 * w_elems, nW);
    cvt_bf16_kernel<<<dim3(nW / (256 * 8)), dim3(256), 0, stream>>>(Wv, Wb + 2 * w_elems, nW);
  }

  dim3 g1(Ntok / 128, Dim / 128, 3);   // (128, 8, 3)
  qkv_gemm_kernel<<<g1, dim3(256), 0, stream>>>(Xb, Wb, Qw, Kw, Vt);

  dim3 g2(Seq / 32, Bsz);              // (64, 8)
  flash_attn_kernel<<<g2, dim3(512), 0, stream>>>(Qw, Kw, Vt, Out);
}
